// Matcher_64768106824201
// MI455X (gfx1250) — matrix operand order repaired and checked by reading, not compiled, not run on hardware
//
#include <hip/hip_runtime.h>

// Problem constants (match reference)
constexpr int B_  = 8;
constexpr int N_  = 256;
constexpr int M_  = 32;
constexpr int HW_ = 32768;

typedef _Float16 v16h __attribute__((ext_vector_type(16)));
typedef _Float16 v4h  __attribute__((ext_vector_type(4)));
typedef float    v8f  __attribute__((ext_vector_type(8)));

// ---------------------------------------------------------------------------
// Kernel 1: convert mask_tgt (f32, binary) -> f16 in workspace, and compute
// per-(b,m) row sums sumT. Grid: B*M blocks x 256 threads.
// ---------------------------------------------------------------------------
__global__ __launch_bounds__(256) void tgt_prep(const float* __restrict__ tgt,
                                                _Float16* __restrict__ tgt16,
                                                float* __restrict__ sumT) {
  const int row = blockIdx.x;        // b*M + m
  const int tid = threadIdx.x;
  const float4* src = (const float4*)(tgt + (size_t)row * HW_);
  v4h* dst = (v4h*)(tgt16 + (size_t)row * HW_);
  float s = 0.f;
  for (int i = tid; i < HW_ / 4; i += 256) {
    float4 v = src[i];
    v4h h;
    h[0] = (_Float16)v.x; h[1] = (_Float16)v.y;
    h[2] = (_Float16)v.z; h[3] = (_Float16)v.w;
    dst[i] = h;
    s += v.x + v.y + v.z + v.w;
  }
  __shared__ float red[256];
  red[tid] = s;
  __syncthreads();
  for (int o = 128; o > 0; o >>= 1) {
    if (tid < o) red[tid] += red[tid + o];
    __syncthreads();
  }
  if (tid == 0) sumT[row] = red[0];
}

// ---------------------------------------------------------------------------
// Kernel 2: main WMMA pass. grid = (B*16 n-tiles, 4 k-slices), 256 threads
// (8 waves of 32). Each wave owns four 16x16 f32 accumulators:
//   accX{0,1}: sum_c x[n,c]*tgt[m,c]          (m-tile 0 / 1)
//   accS{0,1}: sum_c sigmoid(x)[n,c]*tgt[m,c]
// plus per-lane f32 partials of softplus(x) and sigmoid(x) row sums.
//
// Element math, all raw hardware ops (1 v_exp + 1 v_rcp + 1 v_log each):
//   e  = exp(-|x|)                         v_mul(|.|) + v_exp
//   r  = rcp(1+e)                          v_add + v_rcp (~1ulp)
//   sigmoid  = x>=0 ? r : e*r              v_mul + v_cmp + v_cndmask
//   softplus = 0.5*(x+|x|) + ln2*log2(1+e) v_add(|.|) + v_fma + v_log + v_fma
// 1+e in (1,2] -> never subnormal, so the raw v_log_f32 needs no guard.
// ---------------------------------------------------------------------------
__global__ __launch_bounds__(256) void cost_gemm(const float* __restrict__ mh,
                                                 const _Float16* __restrict__ tgt16,
                                                 float* __restrict__ gDotX,
                                                 float* __restrict__ gDotS,
                                                 float* __restrict__ gSP,
                                                 float* __restrict__ gSG) {
  const int tid  = threadIdx.x;
  const int bb   = blockIdx.x >> 4;           // batch
  const int n0   = (blockIdx.x & 15) << 4;    // n-tile base
  const int ks   = blockIdx.y;                // k-slice (0..3)
  const int wave = tid >> 5;
  const int l    = tid & 31;
  const int lm   = l & 15;
  const bool hi  = (l >= 16);

  __shared__ float ldsX[16][32];
  __shared__ float ldsS[16][32];
  __shared__ float ldsSP[16];
  __shared__ float ldsSG[16];
  {
    float* zx = &ldsX[0][0];
    float* zs = &ldsS[0][0];
    zx[tid] = 0.f; zx[tid + 256] = 0.f;
    zs[tid] = 0.f; zs[tid + 256] = 0.f;
    if (tid < 16) { ldsSP[tid] = 0.f; ldsSG[tid] = 0.f; }
  }
  __syncthreads();

  const float* mhb = mh + (size_t)bb * HW_ * N_;   // [HW][N], n contiguous
  // B rows in f16 tgt: lane lm -> m = lm (+16 for tile 1)
  const size_t rowB0 = ((size_t)(bb * M_) + lm) * HW_;
  const size_t rowB1 = rowB0 + (size_t)16 * HW_;
  // A fragment addressing (contiguous k = 16h + e, same order as B):
  //   lo lanes: K = 0..15; hi lanes: K = 16..31
  const int hioffA = hi ? 16 * N_ : 0;  // +16 columns for hi half-wave
  const int hioffB = hi ? 16 : 0;       // B: hi lanes hold K=16..31
  const int nidx   = n0 + lm;

  const v8f zero8 = {0.f, 0.f, 0.f, 0.f, 0.f, 0.f, 0.f, 0.f};
  v8f accX0 = zero8, accX1 = zero8, accS0 = zero8, accS1 = zero8;
  float spAcc = 0.f, sgAcc = 0.f;

  constexpr float LN2 = 0.6931471805599453f;

  const int chunksPerSlice = (HW_ / 32) / 4;   // 256 K-chunks of 32
  for (int i = wave; i < chunksPerSlice; i += 8) {
    const int c0 = (ks * chunksPerSlice + i) * 32;
    const float* a_base = mhb + (size_t)c0 * N_ + hioffA + nidx;

    // Prefetch this wave's next A chunk (c0 + 8*32 columns ahead).
    __builtin_prefetch(a_base + (size_t)(8 * 32) * N_, 0, 1);

    v16h ax, as_;
#pragma unroll
    for (int e = 0; e < 16; ++e) {
      const int ke = e;                                // K within half-group
      const float xv  = a_base[(size_t)ke * N_];
      const float ex  = __expf(-fabsf(xv));            // v_mul(|.|) + v_exp
      const float opx = 1.0f + ex;
      const float r   = __builtin_amdgcn_rcpf(opx);    // raw v_rcp
      const float sg  = (xv >= 0.0f) ? r : ex * r;     // sigmoid(x)
      ax[e]  = (_Float16)xv;
      as_[e] = (_Float16)sg;
      // softplus(x) = max(x,0) + ln(1+e); max(x,0) = 0.5*(x+|x|)
      spAcc = fmaf(0.5f, xv + fabsf(xv), spAcc);
      spAcc = fmaf(LN2, __builtin_amdgcn_logf(opx), spAcc);  // raw v_log
      sgAcc += sg;
    }

    const v16h b0 = *(const v16h*)(tgt16 + rowB0 + c0 + hioffB);
    const v16h b1 = *(const v16h*)(tgt16 + rowB1 + c0 + hioffB);

    accX0 = __builtin_amdgcn_wmma_f32_16x16x32_f16(false, ax,  false, b0, (short)0, accX0, false, false);
    accX1 = __builtin_amdgcn_wmma_f32_16x16x32_f16(false, ax,  false, b1, (short)0, accX1, false, false);
    accS0 = __builtin_amdgcn_wmma_f32_16x16x32_f16(false, as_, false, b0, (short)0, accS0, false, false);
    accS1 = __builtin_amdgcn_wmma_f32_16x16x32_f16(false, as_, false, b1, (short)0, accS1, false, false);
  }

  // Cross-wave reduction in LDS. C/D layout: VGPR r, lo lanes -> row r,
  // hi lanes -> row r+8; column = lane%16 (+16 for m-tile 1).
  const int nlBase = hi ? 8 : 0;
#pragma unroll
  for (int r = 0; r < 8; ++r) {
    atomicAdd(&ldsX[nlBase + r][lm],      accX0[r]);
    atomicAdd(&ldsX[nlBase + r][16 + lm], accX1[r]);
    atomicAdd(&ldsS[nlBase + r][lm],      accS0[r]);
    atomicAdd(&ldsS[nlBase + r][16 + lm], accS1[r]);
  }
  atomicAdd(&ldsSP[lm], spAcc);
  atomicAdd(&ldsSG[lm], sgAcc);
  __syncthreads();

  // Flush block partials to global accumulators (split-K reduction).
  for (int idx = tid; idx < 512; idx += 256) {
    const int nl = idx >> 5, m = idx & 31;
    const size_t g = ((size_t)(bb * N_) + n0 + nl) * M_ + m;
    unsafeAtomicAdd(&gDotX[g], ldsX[nl][m]);
    unsafeAtomicAdd(&gDotS[g], ldsS[nl][m]);
  }
  if (tid < 16) {
    unsafeAtomicAdd(&gSP[bb * N_ + n0 + tid], ldsSP[tid]);
    unsafeAtomicAdd(&gSG[bb * N_ + n0 + tid], ldsSG[tid]);
  }
}

// ---------------------------------------------------------------------------
// Kernel 3: epilogue — combine class / center / mask / dice terms.
// grid = B*N*M/256 blocks x 256 threads, one output element per thread.
// ---------------------------------------------------------------------------
__global__ __launch_bounds__(256) void epilogue(const float* __restrict__ prob,
                                                const float* __restrict__ center,
                                                const int* __restrict__ labels,
                                                const float* __restrict__ gDotX,
                                                const float* __restrict__ gDotS,
                                                const float* __restrict__ gSP,
                                                const float* __restrict__ gSG,
                                                const float* __restrict__ sumT,
                                                float* __restrict__ out) {
  const int gid = blockIdx.x * 256 + threadIdx.x;   // [B,N,M] flat
  const int m  = gid & 31;
  const int n  = (gid >> 5) & 255;
  const int bb = gid >> 13;                         // N*M = 8192

  const float p   = prob[bb * N_ + n];
  const int  lbl  = labels[bb * M_ + m];
  const float cls = -(lbl ? p : (1.0f - p));

  const float dotX = gDotX[gid];
  const float dotS = gDotS[gid];
  const float sp   = gSP[bb * N_ + n];
  const float sg   = gSG[bb * N_ + n];
  const float st   = sumT[bb * M_ + m];

  const float cost_mask = (sp - dotX) * (1.0f / (float)HW_);
  const float cost_dice = 1.0f - (2.0f * dotS + 1.0f) / (sg + st + 1.0f);

  out[gid] = 2.0f * cls + center[gid] + 5.0f * cost_mask + 2.0f * cost_dice;
}

// ---------------------------------------------------------------------------
extern "C" void kernel_launch(void* const* d_in, const int* in_sizes, int n_in,
                              void* d_out, int out_size, void* d_ws, size_t ws_size,
                              hipStream_t stream) {
  const float* prob   = (const float*)d_in[0];   // [B,N,1]
  const float* center = (const float*)d_in[1];   // [B,N,M]
  const float* mh     = (const float*)d_in[2];   // [B,HW,N]
  const float* tgt    = (const float*)d_in[3];   // [B,M,HW]
  const int*   labels = (const int*)d_in[4];     // [B,M]
  float* out = (float*)d_out;

  // Workspace layout
  char* ws = (char*)d_ws;
  _Float16* tgt16 = (_Float16*)ws;                                // 16 MiB
  float* sumT  = (float*)(ws + (size_t)B_ * M_ * HW_ * sizeof(_Float16));
  float* gDotX = sumT  + B_ * M_;
  float* gDotS = gDotX + B_ * N_ * M_;
  float* gSP   = gDotS + B_ * N_ * M_;
  float* gSG   = gSP   + B_ * N_;

  // Zero the atomic accumulation region (gDotX..gSG contiguous).
  const size_t zeroBytes = (size_t)(2 * B_ * N_ * M_ + 2 * B_ * N_) * sizeof(float);
  hipMemsetAsync(gDotX, 0, zeroBytes, stream);

  tgt_prep<<<B_ * M_, 256, 0, stream>>>(tgt, tgt16, sumT);

  dim3 grid(B_ * 16, 4);   // 128 (b, n-tile) x 4 k-slices
  cost_gemm<<<grid, 256, 0, stream>>>(mh, tgt16, gDotX, gDotS, gSP, gSG);

  epilogue<<<(B_ * N_ * M_) / 256, 256, 0, stream>>>(prob, center, labels,
                                                     gDotX, gDotS, gSP, gSG,
                                                     sumT, out);
}
